// SpatialSumLayer_75737453298221
// MI455X (gfx1250) — matrix operand order repaired and checked by reading, not compiled, not run on hardware
//
#include <hip/hip_runtime.h>

typedef __attribute__((ext_vector_type(16))) _Float16 v16h;
typedef __attribute__((ext_vector_type(8)))  _Float16 v8h;
typedef __attribute__((ext_vector_type(8)))  float    v8f;

#define B_  64
#define OC_ 64
#define IC_ 64
#define L_  1024

// ---------------------------------------------------------------------------
// Phase 1: per (row r, l) reduce over IC and pack exp() into f16, transposed.
//   src: [R][IC][L] f32   (R = B for x, R = OC for weight)
//   dst: [L][R][IC] f16   = exp(src - max_ic(src))
//   red: [R][L]    f32    = max_ic (mode 0)  or  log(sum_ic exp(src-max)) (mode 1)
// ---------------------------------------------------------------------------
__global__ __launch_bounds__(256) void pack_exp_kernel(
    const float* __restrict__ src, _Float16* __restrict__ dst,
    float* __restrict__ red, int mode) {
  const int r  = blockIdx.x >> 5;         // 0..63 (b or oc)
  const int l0 = (blockIdx.x & 31) * 32;  // 32 l-columns per block
  __shared__ float tile[IC_][33];         // padded to dodge bank conflicts
  __shared__ float mval[32];
  const int tid = threadIdx.x;
  const int rr = tid >> 5, c = tid & 31;
#pragma unroll
  for (int ib = 0; ib < 8; ++ib) {        // coalesced loads along L
    const int ic = ib * 8 + rr;
    tile[ic][c] = src[((size_t)r * IC_ + ic) * L_ + l0 + c];
  }
  __syncthreads();
  if (tid < 32) {                         // one lane per l-column
    float m = -INFINITY;
#pragma unroll
    for (int ic = 0; ic < IC_; ++ic) m = fmaxf(m, tile[ic][tid]);
    mval[tid] = m;
    float o = m;
    if (mode) {
      float s = 0.f;
#pragma unroll
      for (int ic = 0; ic < IC_; ++ic) s += __expf(tile[ic][tid] - m);
      o = __logf(s);
    }
    red[(size_t)r * L_ + l0 + tid] = o;
  }
  __syncthreads();
  // pack: 8 threads per (l) row, 8 f16 each -> one 16B store, coalesced rows
  const int li = tid >> 3, k8 = (tid & 7) * 8;
  const float m = mval[li];
  v8h pk;
#pragma unroll
  for (int j = 0; j < 8; ++j) pk[j] = (_Float16)__expf(tile[k8 + j][li] - m);
  *(v8h*)(dst + ((size_t)(l0 + li) * B_ + r) * IC_ + k8) = pk;
}

// ---------------------------------------------------------------------------
// Phase 2: per-l 64x64x64 GEMM  S_l = Ex_l (64xIC) * Ew_l^T (ICx64), f16 WMMA.
// Fragment layouts used here (wave32, 16x16x32 f16), contiguous k, not the ISA table order:
//   A 16x32: lanes 0-15 row M, elems K0..15; lanes 16-31 K16..31
//   B 32x16: lane&15 = column N; lanes 0-15 hold K0..15, lanes 16-31 K16..31
// ---------------------------------------------------------------------------
__device__ inline v16h load_a_frag(const _Float16* base, int m0, int k0) {
  const int t = threadIdx.x & 31;
  const _Float16* p = base + (m0 + (t & 15)) * IC_ + k0 + ((t >> 4) << 4);
  v8h lo = *(const v8h*)p;          // K k0+{0..7}   (or +16 for upper lanes)
  v8h hi = *(const v8h*)(p + 8);    // K k0+{8..15}  (or +24)
  return __builtin_shufflevector(lo, hi, 0,1,2,3,4,5,6,7,8,9,10,11,12,13,14,15);
}
__device__ inline v16h load_b_frag(const _Float16* base, int n0, int k0) {
  const int t = threadIdx.x & 31;
  const _Float16* p = base + (n0 + (t & 15)) * IC_ + k0 + ((t >> 4) << 4);
  v8h lo = *(const v8h*)p;          // K k0+{0..7}  (or +16)
  v8h hi = *(const v8h*)(p + 8);    // K k0+{8..15} (or +24)
  return __builtin_shufflevector(lo, hi, 0,1,2,3,4,5,6,7,8,9,10,11,12,13,14,15);
}

__global__ __launch_bounds__(256) void lse_gemm_kernel(
    const _Float16* __restrict__ xp, const _Float16* __restrict__ wp,
    float* __restrict__ out_s) {
  const int l    = blockIdx.x;           // one l per workgroup
  const int wave = threadIdx.x >> 5;     // 8 waves: 4x4 tiles of 16x16
  const int m0   = (wave >> 1) * 16;
  const int n0   = (wave & 1) * 32;      // each wave: tiles (m0,n0) and (m0,n0+16)
  const _Float16* A  = xp + (size_t)l * B_ * IC_;   // [b][ic]
  const _Float16* Bm = wp + (size_t)l * OC_ * IC_;  // [oc][ic]  (== B^T rows)
  v8f c0 = {}, c1 = {};
#pragma unroll
  for (int k0 = 0; k0 < IC_; k0 += 32) {
    v16h a  = load_a_frag(A, m0, k0);
    v16h b0 = load_b_frag(Bm, n0, k0);
    v16h b1 = load_b_frag(Bm, n0 + 16, k0);
    c0 = __builtin_amdgcn_wmma_f32_16x16x32_f16(false, a, false, b0, (short)0, c0, false, false);
    c1 = __builtin_amdgcn_wmma_f32_16x16x32_f16(false, a, false, b1, (short)0, c1, false, false);
  }
  // C/D layout: lane&15 = N col; VGPR v -> M = m0 + v + 8*(lane>=16)
  const int t  = threadIdx.x & 31;
  const int n  = t & 15;
  const int mb = m0 + ((t >> 4) << 3);
  float* orow = out_s + (size_t)l * B_ * OC_;       // [l][b][oc]
#pragma unroll
  for (int v = 0; v < 8; ++v) {
    orow[(size_t)(mb + v) * OC_ + n0 + n]      = c0[v];
    orow[(size_t)(mb + v) * OC_ + n0 + 16 + n] = c1[v];
  }
}

// ---------------------------------------------------------------------------
// Phase 3: epilogue + transpose  out[b][oc][l] = mx[b,l] + log(S[l,b,oc]) - cw[oc,l]
// ---------------------------------------------------------------------------
__global__ __launch_bounds__(256) void epilogue_kernel(
    const float* __restrict__ out_s, const float* __restrict__ mx,
    const float* __restrict__ cw, float* __restrict__ out) {
  const int b  = blockIdx.x >> 5;
  const int l0 = (blockIdx.x & 31) * 32;
  __shared__ float t[OC_][33];
  const int tid = threadIdx.x;
#pragma unroll
  for (int j = 0; j < 8; ++j) {          // coalesced read along oc
    const int e = j * 256 + tid;
    const int li = e >> 6, oc = e & 63;
    const float s = out_s[((size_t)(l0 + li) * B_ + b) * OC_ + oc];
    t[oc][li] = __logf(s) + mx[(size_t)b * L_ + l0 + li] - cw[(size_t)oc * L_ + l0 + li];
  }
  __syncthreads();
#pragma unroll
  for (int j = 0; j < 8; ++j) {          // coalesced write along l
    const int e = j * 256 + tid;
    const int oc = e >> 5, li = e & 31;
    out[((size_t)b * OC_ + oc) * L_ + l0 + li] = t[oc][li];
  }
}

// ---------------------------------------------------------------------------
extern "C" void kernel_launch(void* const* d_in, const int* in_sizes, int n_in,
                              void* d_out, int out_size, void* d_ws, size_t ws_size,
                              hipStream_t stream) {
  (void)in_sizes; (void)n_in; (void)out_size; (void)ws_size;
  const float* x = (const float*)d_in[0];   // [B, IC, L] f32
  const float* w = (const float*)d_in[1];   // [OC, IC, L] f32
  float* out = (float*)d_out;               // [B, OC, L] f32

  char* ws = (char*)d_ws;
  const size_t xp_bytes  = (size_t)L_ * B_  * IC_ * sizeof(_Float16); // 8 MB
  const size_t wp_bytes  = (size_t)L_ * OC_ * IC_ * sizeof(_Float16); // 8 MB
  const size_t red_bytes = (size_t)B_ * L_ * sizeof(float);           // 256 KB
  _Float16* xp    = (_Float16*)ws;
  _Float16* wp    = (_Float16*)(ws + xp_bytes);
  float*    mx    = (float*)(ws + xp_bytes + wp_bytes);
  float*    cw    = (float*)(ws + xp_bytes + wp_bytes + red_bytes);
  float*    out_s = (float*)(ws + xp_bytes + wp_bytes + 2 * red_bytes); // 16 MB

  pack_exp_kernel<<<B_ * (L_ / 32), 256, 0, stream>>>(x, xp, mx, 0);
  pack_exp_kernel<<<OC_ * (L_ / 32), 256, 0, stream>>>(w, wp, cw, 1);
  lse_gemm_kernel<<<L_, 256, 0, stream>>>(xp, wp, out_s);
  epilogue_kernel<<<B_ * (L_ / 32), 256, 0, stream>>>(out_s, mx, cw, out);
}
